// TextSentimentAnalysis_24051816857656
// MI455X (gfx1250) — matrix operand order repaired and checked by reading, not compiled, not run on hardware
//
#include <hip/hip_runtime.h>

// ---------- types ----------
typedef __attribute__((ext_vector_type(16))) __bf16          v16bf;
typedef __attribute__((ext_vector_type(8)))  float           v8f;
typedef __attribute__((ext_vector_type(8)))  unsigned short  v8us;

union FragU { v8us h[2]; v16bf v; };

static __device__ __forceinline__ unsigned short f2bf(float f) {
  unsigned int u = __float_as_uint(f);
  u += 0x7FFFu + ((u >> 16) & 1u);        // round-to-nearest-even
  return (unsigned short)(u >> 16);
}
static __device__ __forceinline__ float sigf(float x) { return 1.0f / (1.0f + __expf(-x)); }
static __device__ __forceinline__ v8f splat8(float x) { v8f r = {x,x,x,x,x,x,x,x}; return r; }

// A fragment: 16x32 bf16 (MxK). ISA layout: lanes 0-15 hold rows 0-15 with
// K in {0..7} U {16..23}; lanes 16-31 hold K in {8..15} U {24..31}.
static __device__ __forceinline__ v16bf load_fragA(const unsigned short* __restrict__ base,
                                                   int row0, int ld, int k0, int lane) {
  int r    = row0 + (lane & 15);
  int half = lane >> 4;
  const unsigned short* p = base + r * ld + k0 + half * 8;
  FragU f;
  f.h[0] = *(const v8us*)(p);
  f.h[1] = *(const v8us*)(p + 16);
  return f.v;
}

// B fragment: 32x16 bf16 (KxN) taken from transposed weights Wt[N,K].
// ISA layout: lane = column (mod 16); lanes 0-15 hold K {0..7} U {16..23}, lanes 16-31 K {8..15} U {24..31}.
static __device__ __forceinline__ v16bf load_fragB(const unsigned short* __restrict__ Wt,
                                                   int n0, int ld, int k0, int lane) {
  int n = n0 + (lane & 15);
  const unsigned short* p = Wt + n * ld + k0 + (lane >> 4) * 8;
  FragU f;
  f.h[0] = *(const v8us*)(p);
  f.h[1] = *(const v8us*)(p + 16);
  return f.v;
}

static __device__ __forceinline__ v8f wmma_bf16(v16bf a, v16bf b, v8f c) {
  return __builtin_amdgcn_wmma_f32_16x16x32_bf16(false, a, false, b, (short)0, c, false, false);
}

// ---------- weight transpose + f32->bf16 ----------
__global__ void transpose_ker(const float* __restrict__ src, unsigned short* __restrict__ dst,
                              int K, int N) {
  long long i = (long long)blockIdx.x * blockDim.x + threadIdx.x;
  long long total = (long long)K * N;
  if (i < total) {
    int k = (int)(i / N), n = (int)(i % N);
    dst[(long long)n * K + k] = f2bf(src[i]);
  }
}

// ---------- embedding gather, f32 table -> bf16 rows ----------
__global__ void embed_ker(const int* __restrict__ tok, const float* __restrict__ emb,
                          unsigned short* __restrict__ out, int ntok, int E) {
  long long i = (long long)blockIdx.x * blockDim.x + threadIdx.x;
  long long total = (long long)ntok * E;
  if (i < total) {
    int ti = (int)(i / E), e = (int)(i % E);
    out[i] = f2bf(emb[(long long)tok[ti] * E + e]);
  }
}

// ---------- persistent fused LSTM layer ----------
// One workgroup (32 wave32s). BN=128 rows. h kept in LDS (bf16), c in registers.
// z = x_t @ W + h @ U + b, gates i,f,g,o; c = sig(f)*c + sig(i)*relu(g); h = sig(o)*relu(c)
template<int E, int H, int S, bool RETSEQ>
__global__ __launch_bounds__(1024)
void lstm_ker(const unsigned short* __restrict__ X,    // [128, S, E] bf16
              const unsigned short* __restrict__ Wt,   // [4H, E]  bf16 (transposed)
              const unsigned short* __restrict__ Ut,   // [4H, H]  bf16 (transposed)
              const float* __restrict__ bias,          // [4H] f32
              unsigned short* __restrict__ Yseq,       // [128, S, H] bf16 (if RETSEQ)
              unsigned short* __restrict__ Ylast) {    // [128, H]    bf16 (if !RETSEQ)
  constexpr int TT  = 8 * (H / 16);   // 16x16 hidden tiles (8 row-blocks x H/16 col-blocks)
  constexpr int TPW = TT / 32;        // tiles per wave (32 waves)
  __shared__ __attribute__((aligned(16))) unsigned short hsh[128 * H];

  const int lane = threadIdx.x & 31;
  const int wv   = threadIdx.x >> 5;

  for (int i = threadIdx.x; i < 128 * H; i += blockDim.x) hsh[i] = 0;

  v8f creg[TPW];
  #pragma unroll
  for (int j = 0; j < TPW; ++j) creg[j] = splat8(0.0f);
  unsigned short hr[TPW][8];

  __syncthreads();

  #pragma unroll 1
  for (int t = 0; t < S; ++t) {
    const unsigned short* Xt = X + t * E;

    #pragma unroll
    for (int j = 0; j < TPW; ++j) {
      const int ti  = wv * TPW + j;
      const int mt  = ti & 7;
      const int nt  = ti >> 3;
      const int col = nt * 16 + (lane & 15);

      if (t + 1 < S)  // warm next timestep's activations into cache
        __builtin_prefetch(Xt + E + (mt * 16 + (lane & 15)) * (S * E), 0, 1);

      v8f zi = splat8(bias[0 * H + col]);
      v8f zf = splat8(bias[1 * H + col]);
      v8f zg = splat8(bias[2 * H + col]);
      v8f zo = splat8(bias[3 * H + col]);

      #pragma unroll 1
      for (int k0 = 0; k0 < E; k0 += 32) {            // x_t @ W  (weights L2-resident)
        v16bf a = load_fragA(Xt, mt * 16, S * E, k0, lane);
        zi = wmma_bf16(a, load_fragB(Wt, 0 * H + nt * 16, E, k0, lane), zi);
        zf = wmma_bf16(a, load_fragB(Wt, 1 * H + nt * 16, E, k0, lane), zf);
        zg = wmma_bf16(a, load_fragB(Wt, 2 * H + nt * 16, E, k0, lane), zg);
        zo = wmma_bf16(a, load_fragB(Wt, 3 * H + nt * 16, E, k0, lane), zo);
      }
      #pragma unroll 1
      for (int k0 = 0; k0 < H; k0 += 32) {            // h @ U  (A from LDS)
        v16bf a = load_fragA(hsh, mt * 16, H, k0, lane);
        zi = wmma_bf16(a, load_fragB(Ut, 0 * H + nt * 16, H, k0, lane), zi);
        zf = wmma_bf16(a, load_fragB(Ut, 1 * H + nt * 16, H, k0, lane), zf);
        zg = wmma_bf16(a, load_fragB(Ut, 2 * H + nt * 16, H, k0, lane), zg);
        zo = wmma_bf16(a, load_fragB(Ut, 3 * H + nt * 16, H, k0, lane), zo);
      }

      #pragma unroll
      for (int e = 0; e < 8; ++e) {
        float iv = sigf(zi[e]);
        float fv = sigf(zf[e]);
        float gv = fmaxf(zg[e], 0.0f);
        float ov = sigf(zo[e]);
        float cv = fv * creg[j][e] + iv * gv;
        creg[j][e] = cv;
        hr[j][e] = f2bf(ov * fmaxf(cv, 0.0f));
      }
    }

    __syncthreads();  // all reads of h(t-1) complete before overwrite

    #pragma unroll
    for (int j = 0; j < TPW; ++j) {
      const int ti  = wv * TPW + j;
      const int mt  = ti & 7;
      const int nt  = ti >> 3;
      const int col = nt * 16 + (lane & 15);
      #pragma unroll
      for (int e = 0; e < 8; ++e) {
        int gr = mt * 16 + ((lane >> 4) << 3) + e;   // C/D layout: row = e + 8*(lane/16)
        hsh[gr * H + col] = hr[j][e];
        if (RETSEQ)            Yseq[(gr * S + t) * H + col] = hr[j][e];
        else if (t == S - 1)   Ylast[gr * H + col]          = hr[j][e];
      }
    }
    __syncthreads();
  }
}

// ---------- dense layer (WMMA), ACT: 0=linear 1=relu 2=sigmoid ----------
template<int IN, int OUT, int M, int ACT>
__global__ __launch_bounds__(256)
void dense_ker(const unsigned short* __restrict__ X,   // [M, IN] bf16
               const unsigned short* __restrict__ Wt,  // [OUT, IN] bf16 (transposed)
               const float* __restrict__ bias,         // [OUT]
               unsigned short* __restrict__ Y,         // [M, OUT] bf16
               float* __restrict__ Yf,                 // optional f32 copy (rows < Mvalid)
               int Mvalid) {
  const int lane = threadIdx.x & 31;
  const int wv   = threadIdx.x >> 5;
  constexpr int MT = M / 16, NT = OUT / 16, TT = MT * NT;

  for (int ti = wv; ti < TT; ti += 8) {
    int mt = ti % MT, nt = ti / MT;
    v8f acc = splat8(bias[nt * 16 + (lane & 15)]);
    #pragma unroll 1
    for (int k0 = 0; k0 < IN; k0 += 32) {
      v16bf a = load_fragA(X, mt * 16, IN, k0, lane);
      v16bf b = load_fragB(Wt, nt * 16, IN, k0, lane);
      acc = wmma_bf16(a, b, acc);
    }
    int col = nt * 16 + (lane & 15);
    #pragma unroll
    for (int e = 0; e < 8; ++e) {
      int gr = mt * 16 + ((lane >> 4) << 3) + e;
      float v = acc[e];
      if (ACT == 1)      v = fmaxf(v, 0.0f);
      else if (ACT == 2) v = sigf(v);
      Y[gr * OUT + col] = f2bf(v);
      if (Yf != nullptr && gr < Mvalid) Yf[gr * OUT + col] = v;
    }
  }
}

// ---------- concat(title,parag) + mean over N=16 -> padded [16,64] bf16 ----------
__global__ void mean_ker(const float* __restrict__ th, const float* __restrict__ ph,
                         unsigned short* __restrict__ xh) {
  int i = threadIdx.x;          // 1024 = 16 rows x 64 cols
  int r = i >> 6, c = i & 63;
  float v = 0.0f;
  if (r < 8) {
    if (c < 32) { for (int n = 0; n < 16; ++n) v += th[(r * 16 + n) * 32 + c]; }
    else        { for (int n = 0; n < 16; ++n) v += ph[(r * 16 + n) * 32 + (c - 32)]; }
    v *= (1.0f / 16.0f);
  }
  xh[i] = f2bf(v);              // rows 8..15 zero-padded for the 16-row WMMA tile
}

// ---------- host launch ----------
extern "C" void kernel_launch(void* const* d_in, const int* in_sizes, int n_in,
                              void* d_out, int out_size, void* d_ws, size_t ws_size,
                              hipStream_t stream) {
  (void)in_sizes; (void)n_in; (void)out_size; (void)ws_size;
  const int*   tok_t = (const int*)  d_in[0];
  const int*   tok_p = (const int*)  d_in[1];
  const float* emb_t = (const float*)d_in[2];
  const float* emb_p = (const float*)d_in[3];
  const float *tl0_W=(const float*)d_in[4],  *tl0_U=(const float*)d_in[5],  *tl0_b=(const float*)d_in[6];
  const float *tl1_W=(const float*)d_in[7],  *tl1_U=(const float*)d_in[8],  *tl1_b=(const float*)d_in[9];
  const float *td0_W=(const float*)d_in[10], *td0_b=(const float*)d_in[11];
  const float *td1_W=(const float*)d_in[12], *td1_b=(const float*)d_in[13];
  const float *td2_W=(const float*)d_in[14], *td2_b=(const float*)d_in[15];
  const float *pl0_W=(const float*)d_in[16], *pl0_U=(const float*)d_in[17], *pl0_b=(const float*)d_in[18];
  const float *pl1_W=(const float*)d_in[19], *pl1_U=(const float*)d_in[20], *pl1_b=(const float*)d_in[21];
  const float *pl2_W=(const float*)d_in[22], *pl2_U=(const float*)d_in[23], *pl2_b=(const float*)d_in[24];
  const float *pd0_W=(const float*)d_in[25], *pd0_b=(const float*)d_in[26];
  const float *pd1_W=(const float*)d_in[27], *pd1_b=(const float*)d_in[28];
  const float *pd2_W=(const float*)d_in[29], *pd2_b=(const float*)d_in[30];
  const float *L0_W =(const float*)d_in[31], *L0_b =(const float*)d_in[32];
  const float *L1_W =(const float*)d_in[33], *L1_b =(const float*)d_in[34];
  const float *L2_W =(const float*)d_in[35], *L2_b =(const float*)d_in[36];
  const float *L3_W =(const float*)d_in[37], *L3_b =(const float*)d_in[38];

  char* wsb = (char*)d_ws; size_t off = 0;
  auto aus = [&](size_t n) -> unsigned short* {
    off = (off + 255) & ~(size_t)255;
    unsigned short* p = (unsigned short*)(wsb + off); off += n * sizeof(unsigned short); return p;
  };
  auto af = [&](size_t n) -> float* {
    off = (off + 255) & ~(size_t)255;
    float* p = (float*)(wsb + off); off += n * sizeof(float); return p;
  };

  // transposed bf16 weights
  unsigned short *tl0Wt=aus(512*128),  *tl0Ut=aus(512*128);
  unsigned short *tl1Wt=aus(256*128),  *tl1Ut=aus(256*64);
  unsigned short *pl0Wt=aus(1024*256), *pl0Ut=aus(1024*256);
  unsigned short *pl1Wt=aus(512*256),  *pl1Ut=aus(512*128);
  unsigned short *pl2Wt=aus(256*128),  *pl2Ut=aus(256*64);
  unsigned short *td0Wt=aus(128*64), *td1Wt=aus(64*128), *td2Wt=aus(32*64);
  unsigned short *pd0Wt=aus(128*64), *pd1Wt=aus(64*128), *pd2Wt=aus(32*64);
  unsigned short *L0Wt=aus(256*64), *L1Wt=aus(128*256), *L2Wt=aus(64*128), *L3Wt=aus(32*64);
  // activations
  unsigned short *xt   = aus((size_t)128*16*128);
  unsigned short *xpe  = aus((size_t)128*512*256);
  unsigned short *tseq = aus((size_t)128*16*128), *tlast = aus(128*64);
  unsigned short *pseq0= aus((size_t)128*512*256);
  unsigned short *pseq1= aus((size_t)128*512*128), *plast = aus(128*64);
  unsigned short *tdh0=aus(128*128), *tdh1=aus(128*64), *tdh2=aus(128*32);
  unsigned short *pdh0=aus(128*128), *pdh1=aus(128*64), *pdh2=aus(128*32);
  float *tdh2f=af(128*32), *pdh2f=af(128*32);
  unsigned short *xh=aus(16*64), *h0=aus(16*256), *h1=aus(16*128), *h2=aus(16*64), *h3=aus(16*32);

  auto T = [&](const float* s, unsigned short* d, int K, int N) {
    long long n = (long long)K * N;
    transpose_ker<<<(int)((n + 255) / 256), 256, 0, stream>>>(s, d, K, N);
  };
  T(tl0_W,tl0Wt,128,512);  T(tl0_U,tl0Ut,128,512);
  T(tl1_W,tl1Wt,128,256);  T(tl1_U,tl1Ut,64,256);
  T(pl0_W,pl0Wt,256,1024); T(pl0_U,pl0Ut,256,1024);
  T(pl1_W,pl1Wt,256,512);  T(pl1_U,pl1Ut,128,512);
  T(pl2_W,pl2Wt,128,256);  T(pl2_U,pl2Ut,64,256);
  T(td0_W,td0Wt,64,128); T(td1_W,td1Wt,128,64); T(td2_W,td2Wt,64,32);
  T(pd0_W,pd0Wt,64,128); T(pd1_W,pd1Wt,128,64); T(pd2_W,pd2Wt,64,32);
  T(L0_W,L0Wt,64,256); T(L1_W,L1Wt,256,128); T(L2_W,L2Wt,128,64); T(L3_W,L3Wt,64,32);

  { long long n = 8192LL * 128;
    embed_ker<<<(int)((n + 255) / 256), 256, 0, stream>>>(tok_t, emb_t, xt, 8192, 128); }
  { long long n = 65536LL * 256;
    embed_ker<<<(int)((n + 255) / 256), 256, 0, stream>>>(tok_p, emb_p, xpe, 65536, 256); }

  // title stack
  lstm_ker<128,128, 16, true ><<<1,1024,0,stream>>>(xt,   tl0Wt, tl0Ut, tl0_b, tseq,  nullptr);
  lstm_ker<128, 64, 16, false><<<1,1024,0,stream>>>(tseq, tl1Wt, tl1Ut, tl1_b, nullptr, tlast);
  // paragraph stack
  lstm_ker<256,256,512, true ><<<1,1024,0,stream>>>(xpe,   pl0Wt, pl0Ut, pl0_b, pseq0, nullptr);
  lstm_ker<256,128,512, true ><<<1,1024,0,stream>>>(pseq0, pl1Wt, pl1Ut, pl1_b, pseq1, nullptr);
  lstm_ker<128, 64,512, false><<<1,1024,0,stream>>>(pseq1, pl2Wt, pl2Ut, pl2_b, nullptr, plast);

  // branch dense heads
  dense_ker< 64,128,128,1><<<1,256,0,stream>>>(tlast, td0Wt, td0_b, tdh0, nullptr, 128);
  dense_ker<128, 64,128,1><<<1,256,0,stream>>>(tdh0,  td1Wt, td1_b, tdh1, nullptr, 128);
  dense_ker< 64, 32,128,0><<<1,256,0,stream>>>(tdh1,  td2Wt, td2_b, tdh2, tdh2f,   128);
  dense_ker< 64,128,128,1><<<1,256,0,stream>>>(plast, pd0Wt, pd0_b, pdh0, nullptr, 128);
  dense_ker<128, 64,128,1><<<1,256,0,stream>>>(pdh0,  pd1Wt, pd1_b, pdh1, nullptr, 128);
  dense_ker< 64, 32,128,0><<<1,256,0,stream>>>(pdh1,  pd2Wt, pd2_b, pdh2, pdh2f,   128);

  mean_ker<<<1,1024,0,stream>>>(tdh2f, pdh2f, xh);

  // head
  dense_ker< 64,256,16,1><<<1,256,0,stream>>>(xh, L0Wt, L0_b, h0, nullptr, 16);
  dense_ker<256,128,16,1><<<1,256,0,stream>>>(h0, L1Wt, L1_b, h1, nullptr, 16);
  dense_ker<128, 64,16,1><<<1,256,0,stream>>>(h1, L2Wt, L2_b, h2, nullptr, 16);
  dense_ker< 64, 32,16,2><<<1,256,0,stream>>>(h2, L3Wt, L3_b, h3, (float*)d_out, 8);
}
